// EncoderBlockOnOnetoken_80865644249600
// MI455X (gfx1250) — matrix operand order repaired and checked by reading, not compiled, not run on hardware
//
#include <hip/hip_runtime.h>
#include <hip/hip_bf16.h>
#include <cstdint>
#include <cstddef>

// ---------------- problem constants ----------------
static constexpr int Bb = 8, Dd = 4, Hh = 64, Ww = 64;
static constexpr int Cc = 256, HEADS = 8, HD = 32, HID = 512;
static constexpr int Nw = 256, INTERVAL = 64, BW = 512;
static constexpr int TOK = Bb * Dd * Hh * Ww;   // 131072 tokens

static constexpr int LDA_S = 40;   // padded LDS row stride (bank-conflict-free, 16B aligned)
static constexpr int LDV_S = 200;  // padded LDS row stride for 192-wide tiles

typedef __attribute__((ext_vector_type(16))) __bf16 v16bf;
typedef __attribute__((ext_vector_type(8)))  __bf16 v8bf;
typedef __attribute__((ext_vector_type(8)))  float  v8f;
typedef __attribute__((__vector_size__(16))) int    vi4;   // builtin's pointee type

// ---------------- CDNA5 async global->LDS (ASYNCcnt) ----------------
#if __has_builtin(__builtin_amdgcn_global_load_async_to_lds_b128)
  #define ASYNC_B128 __builtin_amdgcn_global_load_async_to_lds_b128
  #define USE_ASYNC_LDS 1
#elif __has_builtin(__builtin_amdgcn_global_async_load_to_lds_b128)
  #define ASYNC_B128 __builtin_amdgcn_global_async_load_to_lds_b128
  #define USE_ASYNC_LDS 1
#else
  #define USE_ASYNC_LDS 0
#endif

// AS(3) pointer = low 32 bits of generic LDS address; AS(1) = same 64-bit address.
// Pointee must be v4i (per hipcc diagnostic), non-const.
#define TO_GLOB(p) ((__attribute__((address_space(1))) vi4*)(uintptr_t)(p))
#define TO_LDS(p)  ((__attribute__((address_space(3))) vi4*)(uint32_t)(uintptr_t)(p))

__device__ __forceinline__ void async_cp16(const unsigned short* g, unsigned short* l) {
#if USE_ASYNC_LDS
    ASYNC_B128(TO_GLOB(g), TO_LDS(l), 0, 0);
#else
    *(uint4*)l = *(const uint4*)g;
#endif
}
__device__ __forceinline__ void async_publish_wait() {
#if USE_ASYNC_LDS
  #if __has_builtin(__builtin_amdgcn_s_wait_asynccnt)
    __builtin_amdgcn_s_wait_asynccnt(0);
  #else
    asm volatile("s_wait_asynccnt 0" ::: "memory");
  #endif
#endif
}

__device__ __forceinline__ unsigned short f2bf_u(float f) {
    uint32_t u = __builtin_bit_cast(uint32_t, f);
    uint32_t r = u + 0x7FFFu + ((u >> 16) & 1u);
    return (unsigned short)(r >> 16);
}
__device__ __forceinline__ v8f vzero8() {
    v8f z;
#pragma unroll
    for (int i = 0; i < 8; ++i) z[i] = 0.0f;
    return z;
}
__device__ __forceinline__ v16bf cat16(v8bf lo, v8bf hi) {
    return __builtin_shufflevector(lo, hi, 0, 1, 2, 3, 4, 5, 6, 7,
                                           8, 9, 10, 11, 12, 13, 14, 15);
}

// ---------------- tile staging: A 128x32 + B^T 64x32 per K-slab ----------------
__device__ __forceinline__ void stage_tiles(
    const unsigned short* __restrict__ Ag, int lda,
    const unsigned short* __restrict__ Bt, int ldbt,
    int k0, unsigned short* la, unsigned short* lb, int tid) {
    const int rowA0 = tid >> 2;
    const int rowA1 = rowA0 + 64;
    const int c8    = (tid & 3) << 3;
#if USE_ASYNC_LDS
    async_cp16(Ag + (size_t)rowA0 * lda + k0 + c8, la + rowA0 * LDA_S + c8);
    async_cp16(Ag + (size_t)rowA1 * lda + k0 + c8, la + rowA1 * LDA_S + c8);
    async_cp16(Bt + (size_t)rowA0 * ldbt + k0 + c8, lb + rowA0 * LDA_S + c8);
#else
    uint4 a0 = *(const uint4*)(Ag + (size_t)rowA0 * lda + k0 + c8);
    uint4 a1 = *(const uint4*)(Ag + (size_t)rowA1 * lda + k0 + c8);
    uint4 b0 = *(const uint4*)(Bt + (size_t)rowA0 * ldbt + k0 + c8);
    *(uint4*)(la + rowA0 * LDA_S + c8) = a0;
    *(uint4*)(la + rowA1 * LDA_S + c8) = a1;
    *(uint4*)(lb + rowA0 * LDA_S + c8) = b0;
#endif
}

// ---------------- shared 128x64 WMMA GEMM mainloop (double-buffered) ----------------
// A : bf16 row-major (lda elems/row), rows [0,128) relative to Ag
// Bt: bf16 TRANSPOSED weights (row = output col n, ldbt = K of weight)
// 256 threads = 8 waves; wave w computes rows [16w,16w+16) x 64 cols -> acc[4]
__device__ __forceinline__ void gemm_mainloop(
    const unsigned short* __restrict__ Ag, int lda,
    const unsigned short* __restrict__ Bt, int ldbt,
    int K, v8f acc[4],
    unsigned short* __restrict__ lds_a /*2*128*LDA_S*/,
    unsigned short* __restrict__ lds_bt /*2*64*LDA_S*/) {
    const int tid  = threadIdx.x;
    const int lane = tid & 31;
    const int wv   = tid >> 5;
    const int hf   = (lane >> 4) & 1;
    const int l16  = lane & 15;
    const int nk   = K >> 5;

    stage_tiles(Ag, lda, Bt, ldbt, 0, lds_a, lds_bt, tid);

    for (int i = 0; i < nk; ++i) {
        unsigned short* la = lds_a  + (i & 1) * (128 * LDA_S);
        unsigned short* lb = lds_bt + (i & 1) * (64 * LDA_S);
        async_publish_wait();
        __syncthreads();
        if (i + 1 < nk) {
            unsigned short* la2 = lds_a  + ((i + 1) & 1) * (128 * LDA_S);
            unsigned short* lb2 = lds_bt + ((i + 1) & 1) * (64 * LDA_S);
            stage_tiles(Ag, lda, Bt, ldbt, (i + 1) * 32, la2, lb2, tid);
        }
        // A fragment: m = l16, k = hf*16 + e -> two ds_load_b128 (contiguous)
        const unsigned short* ar = la + (wv * 16 + l16) * LDA_S;
        v16bf af = cat16(*(const v8bf*)(ar + hf * 16),
                         *(const v8bf*)(ar + hf * 16 + 8));
#pragma unroll
        for (int t = 0; t < 4; ++t) {
            // B fragment: n = l16, k = hf*16 + e -> two ds_load_b128 (contiguous)
            const unsigned short* br = lb + (t * 16 + l16) * LDA_S + hf * 16;
            v16bf bfv = cat16(*(const v8bf*)(br), *(const v8bf*)(br + 8));
            acc[t] = __builtin_amdgcn_wmma_f32_16x16x32_bf16(
                false, af, false, bfv, (short)0, acc[t], false, false);
        }
    }
}

// ---------------- kernel: fp32 -> bf16 TRANSPOSED weight convert ----------------
// src: (K, N) row-major fp32 ; dst: (N, K) row-major bf16
__global__ void k_cvt_bf16_t(const float* __restrict__ src,
                             unsigned short* __restrict__ dst, int K, int N) {
    int i = blockIdx.x * blockDim.x + threadIdx.x;
    if (i < K * N) {
        int k = i / N, n = i - k * N;
        dst[(size_t)n * K + k] = f2bf_u(src[i]);
    }
}

// ---------------- kernel: LN1 + window partition -> bf16 ----------------
__global__ void __launch_bounds__(256)
k_ln1_part(const float* __restrict__ x, const float* __restrict__ g,
           const float* __restrict__ b, unsigned short* __restrict__ xw) {
    const int wv = threadIdx.x >> 5, lane = threadIdx.x & 31;
    const int gtok = blockIdx.x * 8 + wv;
    const float* xr = x + (size_t)gtok * Cc;
    float v[8];
    float s = 0.f, ss = 0.f;
#pragma unroll
    for (int i = 0; i < 8; ++i) {
        v[i] = xr[lane * 8 + i];
        s += v[i];
        ss += v[i] * v[i];
    }
#pragma unroll
    for (int m = 1; m < 32; m <<= 1) { s += __shfl_xor(s, m); ss += __shfl_xor(ss, m); }
    const float mean = s * (1.0f / Cc);
    const float var  = ss * (1.0f / Cc) - mean * mean;
    const float rstd = rsqrtf(var + 1e-5f);

    const int bb = gtok >> 14;
    const int dd = (gtok >> 12) & 3;
    const int hh = (gtok >> 6) & 63;
    const int ww = gtok & 63;
    const int wi = bb * 64 + (hh >> 3) * 8 + (ww >> 3);
    const int nn = dd * 64 + (hh & 7) * 8 + (ww & 7);
    unsigned short* orow = xw + ((size_t)wi * Nw + nn) * Cc;
#pragma unroll
    for (int i = 0; i < 8; ++i) {
        int c = lane * 8 + i;
        orow[c] = f2bf_u((v[i] - mean) * rstd * g[c] + b[c]);
    }
}

// ---------------- kernel: fused QKV GEMM (cols: [q(256) | k(256) | v(256)]) ----------------
// q,k stored [wi,head][token][hd] ; v stored TRANSPOSED [wi,head][hd][token]
__global__ void __launch_bounds__(256)
k_qkv(const unsigned short* __restrict__ xw, const unsigned short* __restrict__ wqT,
      const unsigned short* __restrict__ wkvT,
      unsigned short* __restrict__ q, unsigned short* __restrict__ k,
      unsigned short* __restrict__ vv) {
    __shared__ __align__(16) unsigned short lds_a[2 * 128 * LDA_S];
    __shared__ __align__(16) unsigned short lds_bt[2 * 64 * LDA_S];
    const int colBase = blockIdx.x * 64;      // 0..704
    const int rowBase = blockIdx.y * 128;
    const unsigned short* Bt = (colBase < Cc)
        ? (wqT + (size_t)colBase * Cc)
        : (wkvT + (size_t)(colBase - Cc) * Cc);

    v8f acc[4];
#pragma unroll
    for (int t = 0; t < 4; ++t) acc[t] = vzero8();
    gemm_mainloop(xw + (size_t)rowBase * Cc, Cc, Bt, Cc, Cc, acc, lds_a, lds_bt);

    const int lane = threadIdx.x & 31, wv = threadIdx.x >> 5;
    const int hf = lane >> 4, l16 = lane & 15;
    const float scale = 0.17677669529663687f;  // HD^-0.5
#pragma unroll
    for (int t = 0; t < 4; ++t) {
#pragma unroll
        for (int r = 0; r < 8; ++r) {
            const int row = rowBase + wv * 16 + r + 8 * hf;
            const int j   = colBase + t * 16 + l16;
            const int wi = row >> 8, nn = row & 255;
            float val = acc[t][r];
            if (j < Cc) {
                int hd = j & 31, hh = j >> 5;
                q[(((size_t)wi * HEADS + hh) * Nw + nn) * HD + hd] = f2bf_u(val * scale);
            } else if (j < 2 * Cc) {
                int jj = j - Cc; int hd = jj & 31, hh = jj >> 5;
                k[(((size_t)wi * HEADS + hh) * Nw + nn) * HD + hd] = f2bf_u(val);
            } else {
                int jj = j - 2 * Cc; int hd = jj & 31, hh = jj >> 5;
                vv[(((size_t)wi * HEADS + hh) * HD + hd) * Nw + nn] = f2bf_u(val);
            }
        }
    }
}

// ---------------- kernel: windowed attention (one block per window/slab/head) ----------------
__global__ void __launch_bounds__(128)
k_attn(const unsigned short* __restrict__ q, const unsigned short* __restrict__ kk,
       const unsigned short* __restrict__ vT, const float* __restrict__ rpb,
       unsigned short* __restrict__ attn) {
    __shared__ __align__(16) unsigned short lds_p[4 * 16 * LDV_S];  // per-wave P strips
    __shared__ __align__(16) unsigned short lds_vt[32 * LDV_S];     // V^T tile [hd][key]

    const int bid  = blockIdx.x;
    const int head = bid & 7, slab = (bid >> 3) & 3, wi = bid >> 5;
    const int tid = threadIdx.x, lane = tid & 31, wv = tid >> 5;
    const int hf = lane >> 4, l16 = lane & 15;
    const size_t bhBase = ((size_t)wi * HEADS + head) * Nw;        // q/k token rows
    const size_t bvBase = ((size_t)wi * HEADS + head) * HD;        // vT hd rows
    const int skip = slab * INTERVAL;

    // stage V^T (32 hd rows x 192 gathered keys) into LDS, pure 16B chunks (async)
    // 32*192/8 = 768 chunks; key runs are 64-aligned so 8-chunks never straddle
#if USE_ASYNC_LDS
    for (int it = 0; it < 6; ++it) {
        int ch  = tid + it * 128;
        int row = ch / 24;
        int c   = (ch % 24) * 8;
        int tok = c + (c >= skip ? INTERVAL : 0);
        async_cp16(vT + (bvBase + row) * Nw + tok, lds_vt + row * LDV_S + c);
    }
    async_publish_wait();
#else
    uint4 vbuf[6];
    int vrow[6], vc[6];
    for (int it = 0; it < 6; ++it) {
        int ch  = tid + it * 128;
        vrow[it] = ch / 24;
        vc[it]   = (ch % 24) * 8;
        int tok  = vc[it] + (vc[it] >= skip ? INTERVAL : 0);
        vbuf[it] = *(const uint4*)(vT + (bvBase + vrow[it]) * Nw + tok);
    }
    for (int it = 0; it < 6; ++it)
        *(uint4*)(lds_vt + vrow[it] * LDV_S + vc[it]) = vbuf[it];
#endif
    __syncthreads();

    // Q fragment for this wave's 16 query rows
    const int qq0 = wv * 16;
    const unsigned short* qrow = q + (bhBase + skip + qq0 + l16) * HD;
    v16bf qa = cat16(((const v8bf*)qrow)[hf * 2], ((const v8bf*)qrow)[hf * 2 + 1]);

    // scores S = (q*scale) @ k^T over 12 key tiles (K = HD = 32 -> one WMMA each)
    v8f S[12];
#pragma unroll
    for (int ct = 0; ct < 12; ++ct) {
        int j   = ct * 16 + l16;
        int tok = j + (j >= skip ? INTERVAL : 0);
        const v8bf* krow = (const v8bf*)(kk + (bhBase + tok) * HD);
        v16bf kf = cat16(krow[hf * 2], krow[hf * 2 + 1]);
        S[ct] = __builtin_amdgcn_wmma_f32_16x16x32_bf16(
            false, qa, false, kf, (short)0, vzero8(), false, false);
    }

    // relative-position bias (computed analytically)
#pragma unroll
    for (int ct = 0; ct < 12; ++ct) {
        int j   = ct * 16 + l16;
        int tok = j + (j >= skip ? INTERVAL : 0);
        int td = tok >> 6, th = (tok >> 3) & 7, tw = tok & 7;
#pragma unroll
        for (int r = 0; r < 8; ++r) {
            int p  = skip + qq0 + r + 8 * hf;
            int pd = p >> 6, ph = (p >> 3) & 7, pw = p & 7;
            int idx = (pd - td + 3) * 225 + (ph - th + 7) * 15 + (pw - tw + 7);
            S[ct][r] += rpb[idx * HEADS + head];
        }
    }

    // row softmax: rows live on fixed 16-lane halves -> xor-tree width 16
    float inv[8];
#pragma unroll
    for (int r = 0; r < 8; ++r) {
        float m = -3.4e38f;
#pragma unroll
        for (int ct = 0; ct < 12; ++ct) m = fmaxf(m, S[ct][r]);
#pragma unroll
        for (int d = 1; d < 16; d <<= 1) m = fmaxf(m, __shfl_xor(m, d));
        float sum = 0.f;
#pragma unroll
        for (int ct = 0; ct < 12; ++ct) { float p = __expf(S[ct][r] - m); S[ct][r] = p; sum += p; }
#pragma unroll
        for (int d = 1; d < 16; d <<= 1) sum += __shfl_xor(sum, d);
        inv[r] = 1.0f / sum;
    }

    // spill P (bf16) through LDS (C-frag layout -> A-frag layout)
    unsigned short* pr = lds_p + wv * 16 * LDV_S;
#pragma unroll
    for (int ct = 0; ct < 12; ++ct) {
#pragma unroll
        for (int r = 0; r < 8; ++r) {
            pr[(r + 8 * hf) * LDV_S + ct * 16 + l16] = f2bf_u(S[ct][r] * inv[r]);
        }
    }
    __syncthreads();

    // O = P(16x192) @ V(192x32): 6 K-steps x 2 col tiles, all b128 LDS reads
    v8f O[2];
    O[0] = vzero8(); O[1] = vzero8();
#pragma unroll
    for (int kt = 0; kt < 6; ++kt) {
        const unsigned short* ar = pr + l16 * LDV_S + kt * 32;
        v16bf pa = cat16(*(const v8bf*)(ar + hf * 16),
                         *(const v8bf*)(ar + hf * 16 + 8));
#pragma unroll
        for (int t = 0; t < 2; ++t) {
            const unsigned short* br = lds_vt + (t * 16 + l16) * LDV_S + kt * 32 + hf * 16;
            v16bf bfv = cat16(*(const v8bf*)(br), *(const v8bf*)(br + 8));
            O[t] = __builtin_amdgcn_wmma_f32_16x16x32_bf16(
                false, pa, false, bfv, (short)0, O[t], false, false);
        }
    }

    // store context: attn[(wi*N + slab*64 + qq) * C + head*32 + n]
#pragma unroll
    for (int t = 0; t < 2; ++t) {
#pragma unroll
        for (int r = 0; r < 8; ++r) {
            int qq = qq0 + r + 8 * hf;
            size_t off = ((size_t)wi * Nw + skip + qq) * Cc + head * HD + t * 16 + l16;
            attn[off] = f2bf_u(O[t][r]);
        }
    }
}

// ---------------- kernel: output projection + reverse partition + residual ----------------
__global__ void __launch_bounds__(256)
k_proj(const unsigned short* __restrict__ attn, const unsigned short* __restrict__ wpT,
       const float* __restrict__ bp, const float* __restrict__ x,
       float* __restrict__ out) {
    __shared__ __align__(16) unsigned short lds_a[2 * 128 * LDA_S];
    __shared__ __align__(16) unsigned short lds_bt[2 * 64 * LDA_S];
    const int colBase = blockIdx.x * 64;
    const int rowBase = blockIdx.y * 128;
    v8f acc[4];
#pragma unroll
    for (int t = 0; t < 4; ++t) acc[t] = vzero8();
    gemm_mainloop(attn + (size_t)rowBase * Cc, Cc,
                  wpT + (size_t)colBase * Cc, Cc, Cc, acc, lds_a, lds_bt);

    const int lane = threadIdx.x & 31, wv = threadIdx.x >> 5;
    const int hf = lane >> 4, l16 = lane & 15;
#pragma unroll
    for (int t = 0; t < 4; ++t) {
#pragma unroll
        for (int r = 0; r < 8; ++r) {
            const int row = rowBase + wv * 16 + r + 8 * hf;
            const int c   = colBase + t * 16 + l16;
            const int wi = row >> 8, nn = row & 255;
            const int bb = wi >> 6, hblk = (wi >> 3) & 7, wblk = wi & 7;
            const int dd = nn >> 6, hh2 = (nn >> 3) & 7, ww2 = nn & 7;
            const int g = ((bb * Dd + dd) * Hh + hblk * 8 + hh2) * Ww + wblk * 8 + ww2;
            out[(size_t)g * Cc + c] = acc[t][r] + bp[c] + x[(size_t)g * Cc + c];
        }
    }
}

// ---------------- kernel: LN2 (token-major) -> bf16 ----------------
__global__ void __launch_bounds__(256)
k_ln2(const float* __restrict__ x1, const float* __restrict__ g,
      const float* __restrict__ b, unsigned short* __restrict__ y) {
    const int wv = threadIdx.x >> 5, lane = threadIdx.x & 31;
    const int gtok = blockIdx.x * 8 + wv;
    const float* xr = x1 + (size_t)gtok * Cc;
    float v[8];
    float s = 0.f, ss = 0.f;
#pragma unroll
    for (int i = 0; i < 8; ++i) {
        v[i] = xr[lane * 8 + i];
        s += v[i];
        ss += v[i] * v[i];
    }
#pragma unroll
    for (int m = 1; m < 32; m <<= 1) { s += __shfl_xor(s, m); ss += __shfl_xor(ss, m); }
    const float mean = s * (1.0f / Cc);
    const float var  = ss * (1.0f / Cc) - mean * mean;
    const float rstd = rsqrtf(var + 1e-5f);
    unsigned short* orow = y + (size_t)gtok * Cc;
#pragma unroll
    for (int i = 0; i < 8; ++i) {
        int c = lane * 8 + i;
        orow[c] = f2bf_u((v[i] - mean) * rstd * g[c] + b[c]);
    }
}

// ---------------- kernel: FFN layer 1 (C->HID) + bias + exact GELU ----------------
__global__ void __launch_bounds__(256)
k_ffn1(const unsigned short* __restrict__ a, const unsigned short* __restrict__ w1T,
       const float* __restrict__ b1, unsigned short* __restrict__ h) {
    __shared__ __align__(16) unsigned short lds_a[2 * 128 * LDA_S];
    __shared__ __align__(16) unsigned short lds_bt[2 * 64 * LDA_S];
    const int colBase = blockIdx.x * 64;    // 0..448
    const int rowBase = blockIdx.y * 128;
    v8f acc[4];
#pragma unroll
    for (int t = 0; t < 4; ++t) acc[t] = vzero8();
    gemm_mainloop(a + (size_t)rowBase * Cc, Cc,
                  w1T + (size_t)colBase * Cc, Cc, Cc, acc, lds_a, lds_bt);

    const int lane = threadIdx.x & 31, wv = threadIdx.x >> 5;
    const int hf = lane >> 4, l16 = lane & 15;
#pragma unroll
    for (int t = 0; t < 4; ++t) {
#pragma unroll
        for (int r = 0; r < 8; ++r) {
            const int row = rowBase + wv * 16 + r + 8 * hf;
            const int j   = colBase + t * 16 + l16;
            float val = acc[t][r] + b1[j];
            float ge  = 0.5f * val * (1.0f + erff(val * 0.70710678118654752f));
            h[(size_t)row * HID + j] = f2bf_u(ge);
        }
    }
}

// ---------------- kernel: FFN layer 2 (HID->C) + bias + residual ----------------
__global__ void __launch_bounds__(256)
k_ffn2(const unsigned short* __restrict__ h, const unsigned short* __restrict__ w2T,
       const float* __restrict__ b2, float* __restrict__ out) {
    __shared__ __align__(16) unsigned short lds_a[2 * 128 * LDA_S];
    __shared__ __align__(16) unsigned short lds_bt[2 * 64 * LDA_S];
    const int colBase = blockIdx.x * 64;
    const int rowBase = blockIdx.y * 128;
    v8f acc[4];
#pragma unroll
    for (int t = 0; t < 4; ++t) acc[t] = vzero8();
    gemm_mainloop(h + (size_t)rowBase * HID, HID,
                  w2T + (size_t)colBase * HID, HID, HID, acc, lds_a, lds_bt);

    const int lane = threadIdx.x & 31, wv = threadIdx.x >> 5;
    const int hf = lane >> 4, l16 = lane & 15;
#pragma unroll
    for (int t = 0; t < 4; ++t) {
#pragma unroll
        for (int r = 0; r < 8; ++r) {
            const int row = rowBase + wv * 16 + r + 8 * hf;
            const int c   = colBase + t * 16 + l16;
            size_t off = (size_t)row * Cc + c;
            out[off] = acc[t][r] + b2[c] + out[off];
        }
    }
}

// ---------------- launcher ----------------
extern "C" void kernel_launch(void* const* d_in, const int* in_sizes, int n_in,
                              void* d_out, int out_size, void* d_ws, size_t ws_size,
                              hipStream_t stream) {
    (void)in_sizes; (void)n_in; (void)out_size; (void)ws_size;
    const float* x     = (const float*)d_in[0];
    const float* ln1_g = (const float*)d_in[1];
    const float* ln1_b = (const float*)d_in[2];
    const float* ln2_g = (const float*)d_in[3];
    const float* ln2_b = (const float*)d_in[4];
    const float* wq    = (const float*)d_in[5];
    const float* wkv   = (const float*)d_in[6];
    const float* wp    = (const float*)d_in[7];
    const float* bp    = (const float*)d_in[8];
    const float* rpb   = (const float*)d_in[9];
    const float* w1    = (const float*)d_in[10];
    const float* b1    = (const float*)d_in[11];
    const float* w2    = (const float*)d_in[12];
    const float* b2    = (const float*)d_in[13];
    float* out = (float*)d_out;
    char*  ws  = (char*)d_ws;

    // workspace layout (bf16 = unsigned short storage); all weights transposed
    unsigned short* wqT_bf  = (unsigned short*)ws;                // 65536
    unsigned short* wkvT_bf = wqT_bf + 65536;                     // 131072
    unsigned short* wpT_bf  = wkvT_bf + 131072;                   // 65536
    unsigned short* w1T_bf  = wpT_bf + 65536;                     // 131072
    unsigned short* w2T_bf  = w1T_bf + 131072;                    // 131072
    const size_t SZ_BIG = 67108864ull;                            // 64MB region
    unsigned short* xw_bf   = (unsigned short*)(ws + (1ull << 20));
    unsigned short* attn_bf = xw_bf;                              // alias (xw dead after QKV)
    unsigned short* q_bf    = (unsigned short*)(ws + (1ull << 20) + SZ_BIG);
    unsigned short* k_bf    = (unsigned short*)(ws + (1ull << 20) + 2 * SZ_BIG);
    unsigned short* v_bf    = (unsigned short*)(ws + (1ull << 20) + 3 * SZ_BIG);
    unsigned short* h_bf    = q_bf;                               // alias q+k (dead after attn)
    unsigned short* ln2_bf  = v_bf;                               // alias v (dead after attn)

    // weights -> bf16, transposed to (N, K)
    k_cvt_bf16_t<<<256, 256, 0, stream>>>(wq,  wqT_bf,  Cc, Cc);
    k_cvt_bf16_t<<<512, 256, 0, stream>>>(wkv, wkvT_bf, Cc, 2 * Cc);
    k_cvt_bf16_t<<<256, 256, 0, stream>>>(wp,  wpT_bf,  Cc, Cc);
    k_cvt_bf16_t<<<512, 256, 0, stream>>>(w1,  w1T_bf,  Cc, HID);
    k_cvt_bf16_t<<<512, 256, 0, stream>>>(w2,  w2T_bf,  HID, Cc);

    // LN1 + partition
    k_ln1_part<<<TOK / 8, 256, 0, stream>>>(x, ln1_g, ln1_b, xw_bf);
    // fused QKV (cols 768 = q|k|v), v emitted transposed
    k_qkv<<<dim3(12, TOK / 128), 256, 0, stream>>>(xw_bf, wqT_bf, wkvT_bf, q_bf, k_bf, v_bf);
    // attention per (window, slab, head)
    k_attn<<<BW * 4 * HEADS, 128, 0, stream>>>(q_bf, k_bf, v_bf, rpb, attn_bf);
    // projection + reverse + residual -> out holds x1
    k_proj<<<dim3(4, TOK / 128), 256, 0, stream>>>(attn_bf, wpT_bf, bp, x, out);
    // LN2
    k_ln2<<<TOK / 8, 256, 0, stream>>>(out, ln2_g, ln2_b, ln2_bf);
    // FFN
    k_ffn1<<<dim3(8, TOK / 128), 256, 0, stream>>>(ln2_bf, w1T_bf, b1, h_bf);
    k_ffn2<<<dim3(4, TOK / 128), 256, 0, stream>>>(h_bf, w2T_bf, b2, out);
}
